// ProbabilisticGrammar_82514911691070
// MI455X (gfx1250) — matrix operand order repaired and checked by reading, not compiled, not run on hardware
//
#include <hip/hip_runtime.h>
#include <cstdint>

// ---------------------------------------------------------------------------
// Problem constants (match reference setup_inputs)
// ---------------------------------------------------------------------------
#define LAT     256
#define NUNITS  4096
#define NGATES  16384          // 4 * NUNITS
#define NRULES  128
#define NSTEPS  96
#define MAXRHS  2
#define NSYM    8
#define NEGINF  (-1e9f)

#define NB      64             // persistent blocks (grid barrier participants)
#define TPB     256            // 8 waves of 32
#define UPB     (NUNITS / NB)  // 64 units per block

typedef __attribute__((ext_vector_type(16))) __bf16 v16bf;
typedef __attribute__((ext_vector_type(8)))  float  v8f;

union ABPack { uint4 u[2]; v16bf v; };

// ---------------------------------------------------------------------------
// Workspace layout (bytes)
// ---------------------------------------------------------------------------
static const size_t OFF_WT    = 0;                                  // bf16 W^T [NGATES][NUNITS] (128 MB, L2-resident)
static const size_t SZ_WT     = (size_t)NGATES * NUNITS * 2;
static const size_t OFF_GX    = OFF_WT + SZ_WT;                     // fp32 gate_x[NGATES]
static const size_t OFF_C     = OFF_GX + (size_t)NGATES * 4;        // fp32 c[NUNITS]
static const size_t OFF_HB    = OFF_C  + (size_t)NUNITS * 4;        // bf16 h[2][NUNITS]
static const size_t OFF_PART  = OFF_HB + (size_t)2 * NUNITS * 2;    // fp32 partials[NB][NRULES]
static const size_t OFF_STATE = OFF_PART + (size_t)NB * NRULES * 4; // int {sp,symbol,pad,pad,stack[..]}
static const size_t OFF_CNT   = OFF_STATE + 512;                    // u32 counters[2*NSTEPS]
static const size_t WS_NEEDED = OFF_CNT + (size_t)2 * NSTEPS * 4;
static const size_t ZERO_OFF  = OFF_C;
static const size_t ZERO_SZ   = WS_NEEDED - OFF_C;

// ---------------------------------------------------------------------------
// Helpers
// ---------------------------------------------------------------------------
__device__ inline unsigned short f2bf(float f) {       // RNE float -> bf16
  unsigned u = __float_as_uint(f);
  u += 0x7FFFu + ((u >> 16) & 1u);
  return (unsigned short)(u >> 16);
}
__device__ inline float sigm(float x) { return 1.0f / (1.0f + expf(-x)); }

__device__ inline void grid_barrier(unsigned* cnt) {
  __syncthreads();
  if (threadIdx.x == 0) {
    __threadfence();
    __hip_atomic_fetch_add(cnt, 1u, __ATOMIC_ACQ_REL, __HIP_MEMORY_SCOPE_AGENT);
    while (__hip_atomic_load(cnt, __ATOMIC_ACQUIRE, __HIP_MEMORY_SCOPE_AGENT) < (unsigned)NB)
      __builtin_amdgcn_s_sleep(1);
    __threadfence();
  }
  __syncthreads();
}

// CDNA5 async global->LDS copy of 16 bytes (tracked by ASYNCcnt)
__device__ inline void async_copy_b128(unsigned lds_byte_off, unsigned long long gaddr) {
  asm volatile("global_load_async_to_lds_b128 %0, %1, off"
               :: "v"(lds_byte_off), "v"(gaddr) : "memory");
}
__device__ inline void wait_asynccnt0() {
  asm volatile("s_wait_asynccnt 0x0" ::: "memory");
}

// ---------------------------------------------------------------------------
// Precompute kernels
// ---------------------------------------------------------------------------
// Wt[j][k] = bf16( W[(LAT+k)*NGATES + j] )   (recurrent part, transposed)
__global__ void convert_wt_kernel(const float* __restrict__ W,
                                  unsigned short* __restrict__ Wt) {
  size_t idx = (size_t)blockIdx.x * TPB + threadIdx.x;   // NGATES*NUNITS threads
  int k = (int)(idx >> 14);        // 0..4095  (h index)  -- coalesced read in j
  int j = (int)(idx & 16383);      // 0..16383 (gate index)
  Wt[(size_t)j * NUNITS + k] = f2bf(W[(size_t)(LAT + k) * NGATES + j]);
}

// gx[j] = lstm_bias[j] + sum_k x[k] * W[k*NGATES + j]   (exact fp32)
__global__ void gate_x_kernel(const float* __restrict__ x,
                              const float* __restrict__ W,
                              const float* __restrict__ bias,
                              float* __restrict__ gx) {
  int j = blockIdx.x * TPB + threadIdx.x;
  float acc = bias[j];
#pragma unroll 4
  for (int k = 0; k < LAT; ++k)
    acc = fmaf(x[k], W[(size_t)k * NGATES + j], acc);
  gx[j] = acc;
}

__global__ void init_state_kernel(int* __restrict__ state) {
  state[0] = 1;   // sp (stack[0] == START_SYMBOL == 0 from memset)
  state[1] = 0;   // symbol = START_SYMBOL
}

// ---------------------------------------------------------------------------
// Persistent decode kernel: 96 sequential steps, grid-wide barriers
// ---------------------------------------------------------------------------
__global__ __launch_bounds__(TPB, 1)
void grammar_decode_kernel(const unsigned short* __restrict__ Wt,
                           const float* __restrict__ gx,
                           float* __restrict__ cst,
                           unsigned short* __restrict__ hbf,
                           float* __restrict__ part,
                           int* __restrict__ state,
                           unsigned* __restrict__ cnt,
                           const float* __restrict__ dense_w,
                           const float* __restrict__ dense_b,
                           const float* __restrict__ uni,
                           const int* __restrict__ rule_lhs,
                           const int* __restrict__ rhs_table,
                           float* __restrict__ out) {
  const int b    = blockIdx.x;
  const int tid  = threadIdx.x;
  const int lane = tid & 31;
  const int wave = tid >> 5;
  const int m    = lane & 15;       // matrix row within tile
  const int hi   = lane >> 4;       // K-half select (A/B VGPR layout)
  const int ubase = b * UPB;

  __shared__ __align__(32) unsigned short lds_hb[NUNITS]; // staged h (bf16, 8 KB)
  __shared__ float lds_g[4 * UPB];     // gates i|j|f|o for this block's units
  __shared__ float lds_h[UPB];         // h_new for this block's units
  __shared__ float lds_logits[NRULES];

  // raw LDS byte offset of the staging buffer (generic addr low 32 bits)
  const unsigned hb_lds_base = (unsigned)(uintptr_t)(&lds_hb[0]);

  for (int t = 0; t < NSTEPS; ++t) {
    const int sp     = __hip_atomic_load(&state[0], __ATOMIC_ACQUIRE, __HIP_MEMORY_SCOPE_AGENT);
    const int symbol = __hip_atomic_load(&state[1], __ATOMIC_ACQUIRE, __HIP_MEMORY_SCOPE_AGENT);
    const bool valid = sp > 0;
    const int buf = t & 1;

    if (valid) {
      const unsigned short* hb = hbf + (size_t)buf * NUNITS;

      // ---- stage h (8 KB bf16) into LDS with async global->LDS copies ----
      {
        const unsigned loff = hb_lds_base + (unsigned)tid * 32u;
        const unsigned long long ga = (unsigned long long)(uintptr_t)(hb + tid * 16);
        async_copy_b128(loff,        ga);
        async_copy_b128(loff + 16u,  ga + 32u);
      }
      wait_asynccnt0();        // own transfers done (LDS writes landed)
      __syncthreads();         // all waves' transfers visible block-wide

      // ---- gate matvec: 16 tiles of 16 rows, 2 tiles per wave, K=4096 ----
      for (int tt = 0; tt < 2; ++tt) {
        const int tl   = wave * 2 + tt;        // 0..15
        const int g    = tl >> 2;              // gate group i/j/f/o
        const int q    = tl & 3;               // 16-row chunk within 64 units
        const int rowb = g * NUNITS + ubase + q * 16;
        const unsigned short* wrow = Wt + (size_t)(rowb + m) * NUNITS;

        v8f acc = {};
#pragma unroll 2
        for (int kk = 0; kk < NUNITS / 32; ++kk) {
          const int kb   = kk * 32;
          const int aoff = kb + hi * 8;        // A: lanes 0-15 K{0..7,16..23}, 16-31 +8
          ABPack a;
          a.u[0] = *(const uint4*)(wrow + aoff);
          a.u[1] = *(const uint4*)(wrow + aoff + 16);
          __builtin_prefetch((const void*)(wrow + kb + 64), 0, 0);
          const int boff = kb + hi * 8;        // B: h replicated across all 16 cols
          ABPack bm;                           // LDS broadcast reads (ds_load_b128)
          bm.u[0] = *(const uint4*)(&lds_hb[boff]);
          bm.u[1] = *(const uint4*)(&lds_hb[boff + 16]);
          acc = __builtin_amdgcn_wmma_f32_16x16x32_bf16(
              false, a.v, false, bm.v, (short)0, acc, false, false);
        }
        // B columns identical => lane0 holds rows 0..7, lane16 rows 8..15
        if (m == 0) {
          const int lbase = g * UPB + q * 16 + hi * 8;
          const int gbase = rowb + hi * 8;
#pragma unroll
          for (int r = 0; r < 8; ++r)
            lds_g[lbase + r] = acc[r] + gx[gbase + r];
        }
      }
      __syncthreads();

      // ---- LSTM cell for this block's 64 units (TF gate order i,j,f,o) ----
      if (tid < UPB) {
        const float iv = lds_g[tid];
        const float jv = lds_g[UPB + tid];
        const float fv = lds_g[2 * UPB + tid];
        const float ov = lds_g[3 * UPB + tid];
        const float co = cst[ubase + tid];
        const float cn = sigm(fv + 1.0f) * co + sigm(iv) * tanhf(jv);
        const float hn = sigm(ov) * tanhf(cn);
        cst[ubase + tid] = cn;
        hbf[(size_t)(1 - buf) * NUNITS + ubase + tid] = f2bf(hn);
        lds_h[tid] = hn;
      }
      __syncthreads();

      // ---- partial logits over this block's units (coalesced in rule dim) ----
      if (tid < NRULES) {
        float s = 0.0f;
#pragma unroll 4
        for (int u = 0; u < UPB; ++u)
          s = fmaf(lds_h[u], dense_w[(size_t)(ubase + u) * NRULES + tid], s);
        part[b * NRULES + tid] = s;
      }
    }

    grid_barrier(cnt + 2 * t);

    // ---- block 0: deterministic reduce, mask + gumbel, argmax, stack ----
    if (valid && b == 0) {
      if (tid < NRULES) {
        float s = dense_b[tid];
        for (int bb = 0; bb < NB; ++bb)        // fixed order => deterministic
          s += part[bb * NRULES + tid];
        s += (rule_lhs[tid] == symbol) ? 0.0f : NEGINF;
        const float uu  = uni[t * NRULES + tid] + 1e-20f;
        const float gmb = -logf(-logf(uu) + 1e-20f);
        lds_logits[tid] = s + gmb;
      }
      __syncthreads();
      if (tid == 0) {
        float best = lds_logits[0];
        int   arg  = 0;
        for (int r = 1; r < NRULES; ++r)       // strict '>' => first max (jnp.argmax)
          if (lds_logits[r] > best) { best = lds_logits[r]; arg = r; }
        out[t * NRULES + arg] = 1.0f;          // one-hot (rest pre-zeroed)
        int nsp = sp - 1;                      // pop
#pragma unroll
        for (int k2 = 0; k2 < MAXRHS; ++k2) {  // push nonterminals in rhs order
          const int sym = rhs_table[arg * MAXRHS + k2];
          if (sym < NSYM) { state[4 + nsp] = sym; ++nsp; }
        }
        const int nsym = state[4 + (nsp > 0 ? nsp - 1 : 0)];
        __hip_atomic_store(&state[1], nsym, __ATOMIC_RELEASE, __HIP_MEMORY_SCOPE_AGENT);
        __hip_atomic_store(&state[0], nsp, __ATOMIC_RELEASE, __HIP_MEMORY_SCOPE_AGENT);
      }
    }

    grid_barrier(cnt + 2 * t + 1);
  }
}

// ---------------------------------------------------------------------------
// Launch
// ---------------------------------------------------------------------------
extern "C" void kernel_launch(void* const* d_in, const int* in_sizes, int n_in,
                              void* d_out, int out_size, void* d_ws, size_t ws_size,
                              hipStream_t stream) {
  (void)in_sizes; (void)n_in;
  if (ws_size < WS_NEEDED) return;   // workspace too small; nothing safe to do

  const float* x    = (const float*)d_in[0];   // latent_noise   [1,256]
  const float* W    = (const float*)d_in[1];   // lstm_kernel    [4352,16384]
  const float* bias = (const float*)d_in[2];   // lstm_bias      [16384]
  const float* dw   = (const float*)d_in[3];   // dense_w        [4096,128]
  const float* db   = (const float*)d_in[4];   // dense_b        [128]
  const float* uni  = (const float*)d_in[5];   // uniform_noise  [96,128]
  const int*   lhs  = (const int*)d_in[6];     // rule_lhs       [128]
  const int*   rhs  = (const int*)d_in[7];     // rhs_table      [128,2]
  float* out = (float*)d_out;

  char* ws = (char*)d_ws;
  unsigned short* Wt  = (unsigned short*)(ws + OFF_WT);
  float*          gx  = (float*)(ws + OFF_GX);
  float*          cst = (float*)(ws + OFF_C);
  unsigned short* hbf = (unsigned short*)(ws + OFF_HB);
  float*          prt = (float*)(ws + OFF_PART);
  int*            st  = (int*)(ws + OFF_STATE);
  unsigned*       cnt = (unsigned*)(ws + OFF_CNT);

  hipMemsetAsync(ws + ZERO_OFF, 0, ZERO_SZ, stream);            // c, h, state, counters
  hipMemsetAsync(d_out, 0, (size_t)out_size * sizeof(float), stream);

  convert_wt_kernel<<<(NGATES * (size_t)NUNITS) / TPB, TPB, 0, stream>>>(W, Wt);
  gate_x_kernel<<<NGATES / TPB, TPB, 0, stream>>>(x, W, bias, gx);
  init_state_kernel<<<1, 1, 0, stream>>>(st);

  grammar_decode_kernel<<<NB, TPB, 0, stream>>>(
      Wt, gx, cst, hbf, prt, st, cnt, dw, db, uni, lhs, rhs, out);
}
